// GINBlock_309237645712
// MI455X (gfx1250) — matrix operand order repaired and checked by reading, not compiled, not run on hardware
//
#include <hip/hip_runtime.h>

#define NN 100000
#define DD 128
#define EE 3200000
#define TILES 6250  // NN / 16

typedef _Float16 v16h __attribute__((ext_vector_type(16)));
typedef _Float16 v8h  __attribute__((ext_vector_type(8)));
typedef float    v8f  __attribute__((ext_vector_type(8)));
typedef float    v4f  __attribute__((ext_vector_type(4)));
typedef unsigned int u32x4 __attribute__((ext_vector_type(4)));
typedef int          i32x8 __attribute__((ext_vector_type(8)));
typedef int          i32x4 __attribute__((ext_vector_type(4)));

// Guaranteed native no-return HW float atomic (avoids CAS-loop lowering).
__device__ __forceinline__ void gatomic_add_f32(float* p, float v) {
    asm volatile("global_atomic_add_f32 %0, %1, off" : : "v"(p), "v"(v) : "memory");
}

// v16h B/A fragment from halves [0,8) and [16,24) in LDS (two ds_load_b128).
__device__ __forceinline__ v16h ld16h(const _Float16* p) {
    v8h lo = *(const v8h*)(p);
    v8h hi = *(const v8h*)(p + 16);
    return __builtin_shufflevector(lo, hi, 0, 1, 2, 3, 4, 5, 6, 7,
                                   8, 9, 10, 11, 12, 13, 14, 15);
}

// ---------------------------------------------------------------------------
// K1: agg := x   (agg aliases d_out; float4 copy, 3.2M vec4s)
// ---------------------------------------------------------------------------
__global__ void k_copy_x(const v4f* __restrict__ x, v4f* __restrict__ agg) {
    size_t i = (size_t)blockIdx.x * blockDim.x + threadIdx.x;
    agg[i] = x[i];
}

// ---------------------------------------------------------------------------
// K2: zero BN accumulators (256 floats: sums[128], sumsq[128])
// ---------------------------------------------------------------------------
__global__ void k_zero_stats(float* s) { s[threadIdx.x] = 0.0f; }

// ---------------------------------------------------------------------------
// K3: transpose+convert W1,W2 (f32 [k][n]) -> f16 [n][k] for WMMA B-fragments
// w1t and w2t are CONTIGUOUS (64KB total) so one TDM descriptor stages both.
// ---------------------------------------------------------------------------
__global__ void k_prep_w(const float* __restrict__ W1, const float* __restrict__ W2,
                         _Float16* __restrict__ w1t, _Float16* __restrict__ w2t) {
    int id = blockIdx.x * blockDim.x + threadIdx.x;   // 0..32767
    int n = (id >> 7) & 127;
    int k = id & 127;
    if (id < 16384) w1t[n * DD + k] = (_Float16)W1[k * DD + n];
    else            w2t[n * DD + k] = (_Float16)W2[k * DD + n];
}

// ---------------------------------------------------------------------------
// K4: edge scatter-add. One wave per edge; lane l moves features [4l, 4l+4).
// x (51.2MB) and agg (51.2MB) are L2-resident on MI455X (192MB L2).
// ---------------------------------------------------------------------------
__global__ void k_edge_scatter(const float* __restrict__ x,
                               const long long* __restrict__ ei,
                               float* agg) {
    int gid  = blockIdx.x * blockDim.x + threadIdx.x;
    int w    = gid >> 5;
    int lane = threadIdx.x & 31;
    if (w >= EE) return;
    long long src = ei[w];
    long long dst = ei[(size_t)EE + w];
    const v4f* xr = (const v4f*)(x + (size_t)src * DD);
    v4f v = xr[lane];
    float* ar = agg + (size_t)dst * DD + lane * 4;
    gatomic_add_f32(ar + 0, v[0]);
    gatomic_add_f32(ar + 1, v[1]);
    gatomic_add_f32(ar + 2, v[2]);
    gatomic_add_f32(ar + 3, v[3]);
}

// ---------------------------------------------------------------------------
// K5: fused MLP via v_wmma_f32_16x16x32_f16.
// Block = 128 threads = 4 waves; each wave owns one 16-row tile.
//   Stage:  TDM (tensor_load_to_lds) pulls both f16 weight matrices (64KB)
//           into LDS once per block; s_wait_tensorcnt + barrier.
//   GEMM1:  A = agg rows (f32->f16 on load), B = LDS w1; bias+ReLU -> LDS h1.
//   GEMM2:  A = h1 from LDS (A-layout reload), B = LDS w2; bias+ReLU -> out,
//           plus per-channel sum / sumsq atomics for BatchNorm.
// agg aliases out: every out-store is data-dependent on this wave's agg loads.
// ---------------------------------------------------------------------------
__global__ void __launch_bounds__(128)
k_mlp_wmma(const float* agg,
           const _Float16* __restrict__ wgt,   // w1t (32KB) immediately followed by w2t
           const float* __restrict__ b1, const float* __restrict__ b2,
           float* out, float* stats) {
    __shared__ __align__(16) _Float16 wlds[2 * 128 * DD];   // 64 KB: w1 | w2
    __shared__ __align__(16) _Float16 h1s[4][16 * DD];      // 16 KB

    int wv   = threadIdx.x >> 5;
    int lane = threadIdx.x & 31;

    // ---- stage both weight matrices into LDS --------------------------------
#if __has_builtin(__builtin_amdgcn_tensor_load_to_lds)
    if (wv == 0) {   // one wave issues the DMA; EXEC is ignored by TDM
        unsigned lds_base = (unsigned)(uintptr_t)(&wlds[0]);   // low 32b = LDS offset
        unsigned long long ga = (unsigned long long)(uintptr_t)wgt;
        u32x4 g0;
        g0[0] = 1u;                                        // count=1 user D#
        g0[1] = lds_base;                                  // lds_addr
        g0[2] = (unsigned)(ga & 0xFFFFFFFFu);              // global_addr[31:0]
        g0[3] = (unsigned)((ga >> 32) & 0x01FFFFFFu)       // global_addr[56:32]
              | 0x80000000u;                               // type=2 ("image")
        i32x8 g1;
        g1[0] = 0x00010000;        // data_size=1 (2 bytes/elem)
        g1[1] = (int)0x80000000;   // tensor_dim0 = 32768 (bits 79:48 -> low half)
        g1[2] = 0x00010000;        // tensor_dim0 hi = 0, tensor_dim1 = 1
        g1[3] = (int)0x80000000;   // tile_dim0 = 32768 (bits 127:112)
        g1[4] = 0x00000001;        // tile_dim1 = 1, tile_dim2 = 0
        g1[5] = 0x00008000;        // tensor_dim0_stride = 32768
        g1[6] = 0;
        g1[7] = 0;
        i32x4 gz4 = {0, 0, 0, 0};
        i32x8 gz8 = {0, 0, 0, 0, 0, 0, 0, 0};
        // 6-arg variant (clang-23 / therock-10.0 headers):
        // (g0[4xu32], g1[8xi32], g2[4xi32], g3[4xi32], g4[8xi32], cpol)
        __builtin_amdgcn_tensor_load_to_lds(g0, g1, gz4, gz4, gz8, 0);
#if __has_builtin(__builtin_amdgcn_s_wait_tensorcnt)
        __builtin_amdgcn_s_wait_tensorcnt(0);
#else
        asm volatile("s_wait_tensorcnt 0x0" ::: "memory");
#endif
    }
#else
    // Fallback: cooperative 16B-chunk copy (64KB / 16B = 4096 chunks)
    for (int i = threadIdx.x; i < 4096; i += 128)
        ((v8h*)wlds)[i] = ((const v8h*)wgt)[i];
#endif
    __syncthreads();

    int tile = blockIdx.x * 4 + wv;
    if (tile >= TILES) return;            // wave-uniform; EXEC stays all-ones

    int m    = lane & 15;
    int hi   = lane >> 4;                 // half-wave select
    int hi8  = hi << 3;                   // +8  K-offset for A fragments
    int hi16 = hi << 4;                   // unused; B fragments also use hi8

    const float*    arow = agg + (size_t)(tile * 16 + m) * DD;
    const _Float16* wl1  = &wlds[0];
    const _Float16* wl2  = &wlds[128 * DD];

    // ---------------- GEMM1: h1 = relu(agg @ W1 + b1) ----------------
#pragma unroll
    for (int ct = 0; ct < 8; ++ct) {
        v8f c = {};
        int n = ct * 16 + m;
#pragma unroll
        for (int kt = 0; kt < 4; ++kt) {
            int k0 = kt * 32 + hi8;
            v4f f0 = *(const v4f*)(arow + k0);
            v4f f1 = *(const v4f*)(arow + k0 + 4);
            v4f f2 = *(const v4f*)(arow + k0 + 16);
            v4f f3 = *(const v4f*)(arow + k0 + 20);
            v16h a;
#pragma unroll
            for (int i = 0; i < 4; ++i) {
                a[i]      = (_Float16)f0[i];
                a[i + 4]  = (_Float16)f1[i];
                a[i + 8]  = (_Float16)f2[i];
                a[i + 12] = (_Float16)f3[i];
            }
            v16h b = ld16h(wl1 + n * DD + kt * 32 + hi8);
            c = __builtin_amdgcn_wmma_f32_16x16x32_f16(false, a, false, b,
                                                       (short)0, c, false, false);
        }
        float bias = b1[n];
#pragma unroll
        for (int r = 0; r < 8; ++r) {
            float h = c[r] + bias;
            h = h > 0.0f ? h : 0.0f;
            int mc = r + hi8;             // C-layout row
            h1s[wv][mc * DD + n] = (_Float16)h;
        }
    }

    // ---------------- GEMM2: h = relu(h1 @ W2 + b2) ----------------
#pragma unroll
    for (int ct = 0; ct < 8; ++ct) {
        v8f c = {};
        int n = ct * 16 + m;
#pragma unroll
        for (int kt = 0; kt < 4; ++kt) {
            int base = m * DD + kt * 32 + hi8;
            v8h lo = *(const v8h*)(&h1s[wv][base]);
            v8h hh = *(const v8h*)(&h1s[wv][base + 16]);
            v16h a = __builtin_shufflevector(lo, hh, 0, 1, 2, 3, 4, 5, 6, 7,
                                             8, 9, 10, 11, 12, 13, 14, 15);
            v16h b = ld16h(wl2 + n * DD + kt * 32 + hi8);
            c = __builtin_amdgcn_wmma_f32_16x16x32_f16(false, a, false, b,
                                                       (short)0, c, false, false);
        }
        float bias = b2[n];
        float s = 0.0f, s2 = 0.0f;
#pragma unroll
        for (int r = 0; r < 8; ++r) {
            float h = c[r] + bias;
            h = h > 0.0f ? h : 0.0f;
            int mc = r + hi8;
            out[(size_t)(tile * 16 + mc) * DD + n] = h;
            s  += h;
            s2 += h * h;
        }
        atomicAdd(&stats[n], s);
        atomicAdd(&stats[128 + n], s2);
    }
}

// ---------------------------------------------------------------------------
// K6: fold BN stats into per-channel scale/shift: out = h*a[c] + b[c]
// ---------------------------------------------------------------------------
__global__ void k_bn_finalize(const float* __restrict__ stats,
                              const float* __restrict__ gamma,
                              const float* __restrict__ beta,
                              float* __restrict__ ab) {
    int c = threadIdx.x;                  // 128 channels
    float mean = stats[c] * (1.0f / NN);
    float var  = stats[128 + c] * (1.0f / NN) - mean * mean;
    float inv  = rsqrtf(var + 1e-5f);
    float a    = inv * gamma[c];
    ab[c]       = a;
    ab[128 + c] = beta[c] - mean * a;
}

// ---------------------------------------------------------------------------
// K7: apply BN in place over d_out (float4 vectorized)
// ---------------------------------------------------------------------------
__global__ void k_bn_apply(float* out, const float* __restrict__ ab) {
    size_t i = (size_t)blockIdx.x * blockDim.x + threadIdx.x;   // over N*D/4
    int c4 = (int)(i & 31);               // 32 float4s per 128-channel row
    v4f a = ((const v4f*)ab)[c4];
    v4f b = ((const v4f*)(ab + 128))[c4];
    v4f v = ((v4f*)out)[i];
    v4f r;
#pragma unroll
    for (int j = 0; j < 4; ++j) r[j] = v[j] * a[j] + b[j];
    ((v4f*)out)[i] = r;
}

// ---------------------------------------------------------------------------
extern "C" void kernel_launch(void* const* d_in, const int* in_sizes, int n_in,
                              void* d_out, int out_size, void* d_ws, size_t ws_size,
                              hipStream_t stream) {
    const float*     x     = (const float*)d_in[0];
    const long long* ei    = (const long long*)d_in[1];   // int64 [2, E]
    // d_in[2] = edge_attr: unused by the reference computation
    const float*     W1    = (const float*)d_in[3];
    const float*     b1    = (const float*)d_in[4];
    const float*     W2    = (const float*)d_in[5];
    const float*     b2    = (const float*)d_in[6];
    const float*     gamma = (const float*)d_in[7];
    const float*     beta  = (const float*)d_in[8];

    float* out = (float*)d_out;
    float* agg = out;                       // alias: agg lives in d_out

    char* ws = (char*)d_ws;                 // needs ~67KB
    _Float16* w1t  = (_Float16*)(ws);               // 32 KB  \ contiguous 64KB
    _Float16* w2t  = (_Float16*)(ws + 32768);       // 32 KB  / TDM block
    float*    stats = (float*)(ws + 65536);         // 256 floats (sum, sumsq)
    float*    ab    = (float*)(ws + 65536 + 1024);  // 256 floats (scale, shift)

    k_copy_x     <<<(NN * DD / 4) / 256, 256, 0, stream>>>((const v4f*)x, (v4f*)agg);
    k_zero_stats <<<1, 256, 0, stream>>>(stats);
    k_prep_w     <<<32768 / 256, 256, 0, stream>>>(W1, W2, w1t, w2t);
    k_edge_scatter<<<EE / 8, 256, 0, stream>>>(x, ei, agg);   // 8 waves/block
    k_mlp_wmma   <<<(TILES + 3) / 4, 128, 0, stream>>>(agg, w1t, b1, b2, out, stats);
    k_bn_finalize<<<1, 128, 0, stream>>>(stats, gamma, beta, ab);
    k_bn_apply   <<<(NN * DD / 4) / 256, 256, 0, stream>>>(out, ab);
}
